// MultiHeadAttention_49581102465040
// MI455X (gfx1250) — matrix operand order repaired and checked by reading, not compiled, not run on hardware
//
#include <hip/hip_runtime.h>
#include <stdint.h>

// ---------------------------------------------------------------------------
// MHA forward for MI455X (gfx1250, wave32, WMMA bf16 16x16x32, f32 accumulate)
// v3: B-panel staging via GLOBAL_LOAD_ASYNC_TO_LDS_B128 (ASYNCcnt-tracked)
// ---------------------------------------------------------------------------

typedef __attribute__((ext_vector_type(16))) __bf16 v16bf;
typedef __attribute__((ext_vector_type(8)))  __bf16 v8bf;
typedef __attribute__((ext_vector_type(8)))  float  v8f;

static constexpr int Dm = 2048;   // d_model
static constexpr int Hn = 16;     // heads
static constexpr int HD = 128;    // head dim
static constexpr int Tt = 2048;   // seq len

__device__ inline v8f wmma_bf16(v16bf a, v16bf b, v8f c) {
  return __builtin_amdgcn_wmma_f32_16x16x32_bf16(false, a, false, b,
                                                 (short)0, c, false, false);
}

// Async copy of 32B/lane from global (SADDR + VOFF) into LDS (raw address).
// LDS address comes from VDST, global address from SADDR + VADDR; offset adds to both.
__device__ inline void stage_async(uint32_t lds, uint32_t voff, uint64_t sbase) {
  asm volatile(
      "global_load_async_to_lds_b128 %0, %1, %2 offset:0\n\t"
      "global_load_async_to_lds_b128 %0, %1, %2 offset:16"
      :: "v"(lds), "v"(voff), "s"(sbase) : "memory");
}
__device__ inline void wait_async() {
  asm volatile("s_wait_asynccnt 0x0" ::: "memory");
}

// A-frag (16x32 bf16): elems 0..7 at K=half*8.., elems 8..15 at K=16+half*8..
__device__ inline v16bf ldA_f32(const float* __restrict__ p0,
                                const float* __restrict__ p1) {
  v16bf r;
#pragma unroll
  for (int e = 0; e < 8; ++e) { r[e] = (__bf16)p0[e]; r[8 + e] = (__bf16)p1[e]; }
  return r;
}
__device__ inline v16bf ldA_bf16(const __bf16* __restrict__ p0,
                                 const __bf16* __restrict__ p1) {
  v8bf a = *(const v8bf*)p0;
  v8bf b = *(const v8bf*)p1;
  v16bf r;
#pragma unroll
  for (int e = 0; e < 8; ++e) { r[e] = a[e]; r[8 + e] = b[e]; }
  return r;
}
// B-frag (32x16 bf16): lane = column n, elem e -> K = half*16 + e (contiguous)
__device__ inline v16bf ldB_f32(const float* __restrict__ p) {
  v16bf r;
#pragma unroll
  for (int e = 0; e < 16; ++e) r[e] = (__bf16)p[e];
  return r;
}

// ---------------------------------------------------------------------------
// Swizzled fragment layouts (bf16):
//   A-swz:  [mtile16][ks(64)][lane(32)][e(16)]  -> wave loads 1KB contiguous
//   B-swz:  [ntile16][ks(64)][lane(32)][e(16)]  -> wave loads 1KB contiguous
// ---------------------------------------------------------------------------

__global__ __launch_bounds__(256) void conv_x_kernel(
    const float* __restrict__ x, __bf16* __restrict__ Xs) {
  const int tid  = blockIdx.x * 256 + threadIdx.x;   // 256mt*64ks*32lane
  const int lane = tid & 31;
  const int ks   = (tid >> 5) & 63;
  const int mt   = tid >> 11;
  const int half = lane >> 4, l15 = lane & 15;
  const float* p = x + (mt * 16 + l15) * Dm + ks * 32;
  *(v16bf*)(Xs + (size_t)tid * 16) = ldA_f32(p + half * 8, p + 16 + half * 8);
}

__global__ __launch_bounds__(256) void conv_w_kernel(
    const float* __restrict__ W, __bf16* __restrict__ Ws) {
  const int tid  = blockIdx.x * 256 + threadIdx.x;   // nt*64ks*32lane
  const int lane = tid & 31;
  const int ks   = (tid >> 5) & 63;
  const int nt   = tid >> 11;
  const int n = nt * 16 + (lane & 15);
  const int k = ks * 32 + (lane >> 4) * 8;
  *(v16bf*)(Ws + (size_t)tid * 16) = ldA_f32(W + (size_t)n * Dm + k, W + (size_t)n * Dm + k + 16);
}

// ---------------------------------------------------------------------------
// QKV = x @ Wqkv^T + bqkv.  Block = 128(M) x 128(N), 8 waves.
// B panel (8KB/k-step) double-buffered in LDS via async-to-LDS DMA.
// ---------------------------------------------------------------------------
__global__ __launch_bounds__(256) void qkv_gemm(
    const __bf16* __restrict__ Xs, const __bf16* __restrict__ Ws,
    const float* __restrict__ bqkv,
    __bf16* __restrict__ Q, __bf16* __restrict__ Kh, __bf16* __restrict__ Vt) {
  __shared__ __align__(32) __bf16 Bs[2][8][512];
  const int tid = threadIdx.x, lane = tid & 31, w = tid >> 5;
  const int mb = blockIdx.x / 48, nb = blockIdx.x % 48;   // 32 x 48 blocks
  const int mt = mb * 8 + w;
  const int half = lane >> 4, l15 = lane & 15;

  const __bf16* ap = Xs + ((size_t)mt * 64 * 32 + lane) * 16;          // +ks*512
  // LDS raw address: flat LDS pointers carry the byte offset in addr[31:0]
  const uint32_t ldsbase = (uint32_t)(uintptr_t)(&Bs[0][0][0]);
  const uint32_t myoff   = (uint32_t)(w * 1024 + lane * 32);
  const uint64_t sbase   =
      (uint64_t)(uintptr_t)(Ws + (size_t)(nb * 8 + w) * 32768);

  stage_async(ldsbase + myoff, (uint32_t)(lane * 32), sbase);   // ks = 0
  wait_async();
  __syncthreads();

  v8f acc[8] = {};
  for (int ks = 0; ks < 64; ++ks) {
    const int buf = ks & 1;
    if (ks + 1 < 64)
      stage_async(ldsbase + (uint32_t)((buf ^ 1) * 8192) + myoff,
                  (uint32_t)((ks + 1) * 1024 + lane * 32), sbase);
    __builtin_prefetch(ap + (ks + 8) * 512, 0, 0);
    v16bf a = *(const v16bf*)(ap + ks * 512);
#pragma unroll
    for (int j = 0; j < 8; ++j)
      acc[j] = wmma_bf16(a, *(const v16bf*)(&Bs[buf][j][lane * 16]), acc[j]);
    if (ks + 1 < 64) wait_async();
    __syncthreads();
  }

  const int rowbase = half * 8;
#pragma unroll
  for (int j = 0; j < 8; ++j) {
    const int f = nb * 128 + j * 16 + l15;
    const float bias = bqkv[f];
    const int which = f >> 11;          // 0=q 1=k 2=v
    const int fh = f & 2047;
    const int h = fh >> 7, d = fh & 127;
#pragma unroll
    for (int i = 0; i < 8; ++i) {
      const int row = mt * 16 + rowbase + i;
      const int b = row >> 11, t = row & 2047;
      const float v = acc[j][i] + bias;
      if (which == 0)       Q[((b * Hn + h) * Tt + t) * HD + d] = (__bf16)v;
      else if (which == 1)  Kh[((b * Hn + h) * Tt + t) * HD + d] = (__bf16)v;
      else                  Vt[((b * Hn + h) * HD + d) * Tt + t] = (__bf16)v;
    }
  }
}

// ---------------------------------------------------------------------------
// In-place RoPE on Q and K (bf16). One thread per rotation pair.
// ---------------------------------------------------------------------------
__global__ __launch_bounds__(256) void rope_kernel(__bf16* __restrict__ Q,
                                                   __bf16* __restrict__ Kh) {
  const int idx = blockIdx.x * 256 + threadIdx.x;   // B*H*T*64 total
  const int j  = idx & 63;
  const int t  = (idx >> 6) & (Tt - 1);
  const int bh = idx >> 17;
  const float ang = (float)t * __expf(-0.14391157f * (float)j); // ln(1e4)/64
  float s, c;
  __sincosf(ang, &s, &c);
  const int base = (bh * Tt + t) * HD + 2 * j;
  {
    float a = (float)Q[base], b = (float)Q[base + 1];
    Q[base]     = (__bf16)(a * c - b * s);
    Q[base + 1] = (__bf16)(b * c + a * s);
  }
  {
    float a = (float)Kh[base], b = (float)Kh[base + 1];
    Kh[base]     = (__bf16)(a * c - b * s);
    Kh[base + 1] = (__bf16)(b * c + a * s);
  }
}

// ---------------------------------------------------------------------------
// Causal flash attention. One wave per (b,h,16-query tile).
// 32 keys/step: 8 WMMA for S=QK^T, online softmax (shfl row reductions),
// P transposed C->A layout via per-wave LDS bounce, 8 WMMA for P@V.
// Output written directly in A-swizzled layout for the projection GEMM.
// ---------------------------------------------------------------------------
__global__ __launch_bounds__(256) void attn_kernel(
    const __bf16* __restrict__ Q, const __bf16* __restrict__ Kh,
    const __bf16* __restrict__ Vt, __bf16* __restrict__ Os) {
  __shared__ __align__(32) __bf16 plds[8][16][32];   // 1KB/wave P bounce
  const int lane = threadIdx.x & 31, w = threadIdx.x >> 5;
  const int wid = blockIdx.x * 8 + w;   // 0..4095
  const int qt = wid & 127;             // 128 query tiles
  const int bh = wid >> 7;              // 0..31
  const int half = lane >> 4, l15 = lane & 15, rowbase = half * 8;
  const int q0 = qt * 16;

  const __bf16* Qb = Q  + bh * Tt * HD;
  const __bf16* Kb = Kh + bh * Tt * HD;
  const __bf16* Vb = Vt + bh * HD * Tt;

  v16bf qa[4];
#pragma unroll
  for (int s = 0; s < 4; ++s) {
    const __bf16* qp = Qb + (q0 + l15) * HD + s * 32;
    qa[s] = ldA_bf16(qp + half * 16, qp + half * 16 + 8);
  }

  v8f oacc[8] = {};
  float mrow[8], lrow[8];
#pragma unroll
  for (int i = 0; i < 8; ++i) { mrow[i] = -1e30f; lrow[i] = 0.0f; }

  const float scale = 0.08838834764f;   // 1/sqrt(128)
  const int nkt = (q0 + 47) >> 5;

  for (int kt = 0; kt < nkt; ++kt) {
    const int key0 = kt * 32 + l15;
    v8f c0 = {}, c1 = {};
#pragma unroll
    for (int s = 0; s < 4; ++s) {
      c0 = wmma_bf16(qa[s], *(const v16bf*)(Kb + key0 * HD + s * 32 + half * 16), c0);
      c1 = wmma_bf16(qa[s], *(const v16bf*)(Kb + (key0 + 16) * HD + s * 32 + half * 16), c1);
    }

    const bool need_mask = (kt * 32 + 31) > q0;
    float fsc[8];
#pragma unroll
    for (int i = 0; i < 8; ++i) {
      float s0 = c0[i] * scale, s1 = c1[i] * scale;
      if (need_mask) {
        const int qrow = q0 + rowbase + i;
        if (key0 > qrow)      s0 = -1e30f;
        if (key0 + 16 > qrow) s1 = -1e30f;
      }
      float mt_ = fmaxf(s0, s1);
      mt_ = fmaxf(mt_, __shfl_xor(mt_, 1, 32));
      mt_ = fmaxf(mt_, __shfl_xor(mt_, 2, 32));
      mt_ = fmaxf(mt_, __shfl_xor(mt_, 4, 32));
      mt_ = fmaxf(mt_, __shfl_xor(mt_, 8, 32));
      const float mn = fmaxf(mrow[i], mt_);
      const float f  = __expf(mrow[i] - mn);
      const float p0 = __expf(s0 - mn);
      const float p1 = __expf(s1 - mn);
      float rs = p0 + p1;
      rs += __shfl_xor(rs, 1, 32);
      rs += __shfl_xor(rs, 2, 32);
      rs += __shfl_xor(rs, 4, 32);
      rs += __shfl_xor(rs, 8, 32);
      lrow[i] = lrow[i] * f + rs;
      mrow[i] = mn;
      fsc[i]  = f;
      c0[i] = p0; c1[i] = p1;
    }
#pragma unroll
    for (int j = 0; j < 8; ++j)
#pragma unroll
      for (int i = 0; i < 8; ++i) oacc[j][i] *= fsc[i];

    __bf16* pw = &plds[w][0][0];
#pragma unroll
    for (int i = 0; i < 8; ++i) {
      pw[(rowbase + i) * 32 + l15]      = (__bf16)c0[i];
      pw[(rowbase + i) * 32 + 16 + l15] = (__bf16)c1[i];
    }
    __asm__ volatile("s_wait_dscnt 0x0" ::: "memory");  // same-wave RAW on LDS
    const __bf16* pr = pw + l15 * 32;
    v16bf pa = ldA_bf16(pr + half * 16, pr + half * 16 + 8);

#pragma unroll
    for (int j = 0; j < 8; ++j) {
      v16bf vb = *(const v16bf*)(Vb + (j * 16 + l15) * Tt + kt * 32 + half * 16);
      oacc[j] = wmma_bf16(pa, vb, oacc[j]);
    }
  }

  // Epilogue: write A-swizzled O so the projection GEMM streams it coalesced.
  const int b = bh >> 4, h = bh & 15;
  const int mt = (b * Tt + q0) >> 4;
#pragma unroll
  for (int j = 0; j < 8; ++j) {
    const int ks = h * 4 + (j >> 1);
    const int e  = (l15 & 7) + (j & 1) * 8;
    const int lanehalf = (l15 >> 3) & 1;
#pragma unroll
    for (int i = 0; i < 8; ++i) {
      const int lp = lanehalf * 16 + rowbase + i;
      Os[(((size_t)mt * 64 + ks) * 32 + lp) * 16 + e] =
          (__bf16)(oacc[j][i] / lrow[i]);
    }
  }
}

// ---------------------------------------------------------------------------
// out = O @ Wo^T + bo (f32 out). Same async-staged structure as qkv_gemm.
// ---------------------------------------------------------------------------
__global__ __launch_bounds__(256) void proj_gemm(
    const __bf16* __restrict__ Osz, const __bf16* __restrict__ Ws,
    const float* __restrict__ bo, float* __restrict__ out) {
  __shared__ __align__(32) __bf16 Bs[2][8][512];
  const int tid = threadIdx.x, lane = tid & 31, w = tid >> 5;
  const int mb = blockIdx.x / 16, nb = blockIdx.x % 16;   // 32 x 16 blocks
  const int mt = mb * 8 + w;
  const int half = lane >> 4, l15 = lane & 15;

  const __bf16* ap = Osz + ((size_t)mt * 64 * 32 + lane) * 16;
  const uint32_t ldsbase = (uint32_t)(uintptr_t)(&Bs[0][0][0]);
  const uint32_t myoff   = (uint32_t)(w * 1024 + lane * 32);
  const uint64_t sbase   =
      (uint64_t)(uintptr_t)(Ws + (size_t)(nb * 8 + w) * 32768);

  stage_async(ldsbase + myoff, (uint32_t)(lane * 32), sbase);
  wait_async();
  __syncthreads();

  v8f acc[8] = {};
  for (int ks = 0; ks < 64; ++ks) {
    const int buf = ks & 1;
    if (ks + 1 < 64)
      stage_async(ldsbase + (uint32_t)((buf ^ 1) * 8192) + myoff,
                  (uint32_t)((ks + 1) * 1024 + lane * 32), sbase);
    __builtin_prefetch(ap + (ks + 8) * 512, 0, 0);
    v16bf a = *(const v16bf*)(ap + ks * 512);
#pragma unroll
    for (int j = 0; j < 8; ++j)
      acc[j] = wmma_bf16(a, *(const v16bf*)(&Bs[buf][j][lane * 16]), acc[j]);
    if (ks + 1 < 64) wait_async();
    __syncthreads();
  }

  const int rowbase = half * 8;
#pragma unroll
  for (int j = 0; j < 8; ++j) {
    const int n = nb * 128 + j * 16 + l15;
    const float bias = bo[n];
#pragma unroll
    for (int i = 0; i < 8; ++i) {
      const int row = mt * 16 + rowbase + i;
      out[(size_t)row * Dm + n] = acc[j][i] + bias;
    }
  }
}

// ---------------------------------------------------------------------------
extern "C" void kernel_launch(void* const* d_in, const int* in_sizes, int n_in,
                              void* d_out, int out_size, void* d_ws,
                              size_t ws_size, hipStream_t stream) {
  const float* x    = (const float*)d_in[0];
  const float* Wqkv = (const float*)d_in[1];
  const float* bqkv = (const float*)d_in[2];
  const float* Wo   = (const float*)d_in[3];
  const float* bo   = (const float*)d_in[4];
  float* out        = (float*)d_out;

  const size_t MB = 1ull << 20;
  char* ws = (char*)d_ws;
  __bf16* Xs  = (__bf16*)(ws);             // 16 MB  A-swizzled x
  __bf16* Wqs = (__bf16*)(ws + 16 * MB);   // 24 MB  B-swizzled Wqkv
  __bf16* Wos = (__bf16*)(ws + 40 * MB);   //  8 MB  B-swizzled Wo
  __bf16* Q   = (__bf16*)(ws + 48 * MB);   // 16 MB
  __bf16* Kh  = (__bf16*)(ws + 64 * MB);   // 16 MB
  __bf16* Vt  = (__bf16*)(ws + 80 * MB);   // 16 MB  V transposed
  __bf16* Osz = (__bf16*)(ws + 96 * MB);   // 16 MB  A-swizzled attn out

  conv_x_kernel<<<2048, 256, 0, stream>>>(x, Xs);
  conv_w_kernel<<<3072, 256, 0, stream>>>(Wqkv, Wqs);
  conv_w_kernel<<<1024, 256, 0, stream>>>(Wo, Wos);
  qkv_gemm<<<32 * 48, 256, 0, stream>>>(Xs, Wqs, bqkv, Q, Kh, Vt);
  rope_kernel<<<16384, 256, 0, stream>>>(Q, Kh);
  attn_kernel<<<512, 256, 0, stream>>>(Q, Kh, Vt, Osz);
  proj_gemm<<<32 * 16, 256, 0, stream>>>(Osz, Wos, bo, out);
}
